// SW_MetaPath2Vec_86612310491672
// MI455X (gfx1250) — matrix operand order repaired and checked by reading, not compiled, not run on hardware
//
#include <hip/hip_runtime.h>
#include <hip/hip_bf16.h>
#include <math.h>

// Problem constants from the reference
#define NN    2000   // N = N_SRC + N_DST
#define PADN  2016   // padded: 126*16 = 63*32 = 42*48
#define NSRC  1000
#define DD    128    // embedding dim
#define HH    256    // hidden dim
#define BB    100000 // skipgram batch
#define KNEG  5
#define ENUM  200000 // edges per score set

typedef __attribute__((ext_vector_type(16))) _Float16 v16h;
typedef __attribute__((ext_vector_type(8)))  _Float16 v8h;
typedef __attribute__((ext_vector_type(8)))  float    v8f;

// WGP-scope (scope 0) prefetch: pulls into ALL cache levels on miss, unlike
// __builtin_prefetch which only reaches SE/SYS scope (GL2-only, useless here
// since the data is already L2-resident). Scope modifier omitted => scope 0.
__device__ __forceinline__ void prefetch_wgp(const void* p) {
    asm volatile("global_prefetch_b8 %0, off offset:512" :: "v"(p) : "memory");
}

// ---------------------------------------------------------------------------
// Fragment load helpers (wave32 WMMA 16x16x32 f16 layouts, ISA 7.12.2)
// ---------------------------------------------------------------------------
__device__ __forceinline__ v16h load_a_frag(const _Float16* p) {
    // p = &A[row, kBlock + (halfSel<<3)]; lanes 0-15: K {0..7,16..23}, 16-31: {8..15,24..31}
    v8h alo = *(const v8h*)(p);
    v8h ahi = *(const v8h*)(p + 16);
    v16h av;
#pragma unroll
    for (int i = 0; i < 8; ++i) { av[i] = alo[i]; av[i + 8] = ahi[i]; }
    return av;
}
__device__ __forceinline__ v16h load_b_frag(const _Float16* p) {
    // p = &BT[col, kBlock + (halfSel<<3)]; lanes 0-15: K {0..7,16..23}, 16-31: {8..15,24..31}
    v8h blo = *(const v8h*)(p);
    v8h bhi = *(const v8h*)(p + 16);
    v16h bv;
#pragma unroll
    for (int i = 0; i < 8; ++i) { bv[i] = blo[i]; bv[i + 8] = bhi[i]; }
    return bv;
}

// ---------------------------------------------------------------------------
// Register-blocked WMMA GEMM: each wave computes a 48x48 tile (3x3 WMMA tiles).
// C (MxN f32) = A (MxK f16 row-major) @ B, with B given as BT (NxK row-major).
// M, N multiples of 48; K multiple of 32.
// 6 fragment loads feed 9 WMMAs per K-step (3x A-reuse and B-reuse).
// ---------------------------------------------------------------------------
__global__ __launch_bounds__(256) void wmma_gemm48(
    const _Float16* __restrict__ A, const _Float16* __restrict__ BT,
    float* __restrict__ C, int M, int N, int K, int lda, int ldb, int ldc)
{
    const int wave = (int)((blockIdx.x * blockDim.x + threadIdx.x) >> 5);
    const int lane = (int)(threadIdx.x & 31);
    const int gN = N / 48;
    const int gM = M / 48;
    if (wave >= gM * gN) return;           // uniform per-wave exit; EXEC stays full
    const int tm = (wave / gN) * 48;
    const int tn = (wave % gN) * 48;

    const int halfSel = lane >> 4;
    const int l15 = lane & 15;

    const _Float16* aptr[3];
    const _Float16* bptr[3];
#pragma unroll
    for (int i = 0; i < 3; ++i) {
        aptr[i] = A  + (size_t)(tm + i * 16 + l15) * lda + (halfSel << 3);
        bptr[i] = BT + (size_t)(tn + i * 16 + l15) * ldb + (halfSel << 3);
    }

    v8f acc[3][3];
#pragma unroll
    for (int i = 0; i < 3; ++i)
#pragma unroll
        for (int j = 0; j < 3; ++j) acc[i][j] = (v8f){};

    for (int kk = 0; kk < K; kk += 32) {
        // Prefetch the next stretch of the sequentially-consumed K rows into
        // the WGP cache (scope 0).
        prefetch_wgp(aptr[0] + kk);
        prefetch_wgp(bptr[0] + kk);

        v16h av[3], bv[3];
#pragma unroll
        for (int i = 0; i < 3; ++i) av[i] = load_a_frag(aptr[i] + kk);
#pragma unroll
        for (int j = 0; j < 3; ++j) bv[j] = load_b_frag(bptr[j] + kk);

#pragma unroll
        for (int i = 0; i < 3; ++i)
#pragma unroll
            for (int j = 0; j < 3; ++j)
                acc[i][j] = __builtin_amdgcn_wmma_f32_16x16x32_f16(
                    false, av[i], false, bv[j], (short)0, acc[i][j], false, false);
    }

    const int rbase = tm + (halfSel << 3);
#pragma unroll
    for (int i = 0; i < 3; ++i) {
#pragma unroll
        for (int j = 0; j < 3; ++j) {
            const int row0 = rbase + i * 16;
            const int col  = tn + j * 16 + l15;
#pragma unroll
            for (int v = 0; v < 8; ++v)
                C[(size_t)(row0 + v) * ldc + col] = acc[i][j][v];
        }
    }
}

// ---------------------------------------------------------------------------
// Simple WMMA GEMM (one 16x16 tile per wave) for the small GEMMs.
// ---------------------------------------------------------------------------
__global__ __launch_bounds__(256) void wmma_gemm_f16f32(
    const _Float16* __restrict__ A, const _Float16* __restrict__ BT,
    float* __restrict__ C, int M, int N, int K, int lda, int ldb, int ldc)
{
    const int wave = (int)((blockIdx.x * blockDim.x + threadIdx.x) >> 5);
    const int lane = (int)(threadIdx.x & 31);
    const int tilesN = N >> 4;
    const int tilesM = M >> 4;
    if (wave >= tilesM * tilesN) return;
    const int tm = (wave / tilesN) << 4;
    const int tn = (wave % tilesN) << 4;

    const int halfSel = lane >> 4;
    const _Float16* aptr = A  + (size_t)(tm + (lane & 15)) * lda + (halfSel << 3);
    const _Float16* bptr = BT + (size_t)(tn + (lane & 15)) * ldb + (halfSel << 3);

    v8f acc = {};
    for (int kk = 0; kk < K; kk += 32) {
        v16h av = load_a_frag(aptr + kk);
        v16h bv = load_b_frag(bptr + kk);
        acc = __builtin_amdgcn_wmma_f32_16x16x32_f16(
            false, av, false, bv, (short)0, acc, false, false);
    }
    const int row0 = tm + (halfSel << 3);
    const int col  = tn + (lane & 15);
#pragma unroll
    for (int v = 0; v < 8; ++v)
        C[(size_t)(row0 + v) * ldc + col] = acc[v];
}

// ---------------------------------------------------------------------------
// f32 -> padded f16 (row-major copy, zero pad outside valid region)
// ---------------------------------------------------------------------------
__global__ void f32_to_f16_pad(_Float16* __restrict__ dst, const float* __restrict__ src,
                               int validRows, int validCols, int ldsrc,
                               int dstRows, int lddst)
{
    size_t i = (size_t)blockIdx.x * blockDim.x + threadIdx.x;
    size_t total = (size_t)dstRows * lddst;
    if (i >= total) return;
    int r = (int)(i / lddst), c = (int)(i % lddst);
    float v = (r < validRows && c < validCols) ? src[(size_t)r * ldsrc + c] : 0.0f;
    dst[i] = (_Float16)v;
}

// f32 -> padded f16 with transpose: dst[r,c] = src[c,r] (src is srcRows x srcCols)
__global__ void f32T_to_f16_pad(_Float16* __restrict__ dst, const float* __restrict__ src,
                                int dstRows, int dstCols, int srcRows, int srcCols)
{
    size_t i = (size_t)blockIdx.x * blockDim.x + threadIdx.x;
    size_t total = (size_t)dstRows * dstCols;
    if (i >= total) return;
    int r = (int)(i / dstCols), c = (int)(i % dstCols);
    float v = (c < srcRows && r < srcCols) ? src[(size_t)c * srcCols + r] : 0.0f;
    dst[i] = (_Float16)v;
}

__global__ void diag_kernel(float* __restrict__ d, const float* __restrict__ A2f)
{
    int i = blockIdx.x * blockDim.x + threadIdx.x;
    if (i >= PADN) return;
    d[i] = (i < NN) ? A2f[(size_t)i * PADN + i] : 0.0f;
}

// In place: Pf <- gamma_raw = ok ? (Pf*adj)/(d_i*d_j) : 0   (zero on pad)
__global__ void gamma_raw(float* __restrict__ Pf, const float* __restrict__ adj,
                          const float* __restrict__ d)
{
    size_t i = (size_t)blockIdx.x * blockDim.x + threadIdx.x;
    if (i >= (size_t)PADN * PADN) return;
    int r = (int)(i / PADN), c = (int)(i % PADN);
    float g = 0.0f;
    if (r < NN && c < NN) {
        float p1 = Pf[i] * adj[(size_t)r * NN + c];
        float p2 = d[r] * d[c];
        g = (p2 > 0.0f) ? (p1 / p2) : 0.0f;
    }
    Pf[i] = g;
}

__global__ __launch_bounds__(256) void row_norm(float* __restrict__ norms,
                                                const float* __restrict__ G)
{
    int row = blockIdx.x;
    float s = 0.0f;
    for (int c = threadIdx.x; c < PADN; c += blockDim.x) {
        float g = G[(size_t)row * PADN + c];
        s += g * g;
    }
    __shared__ float buf[256];
    buf[threadIdx.x] = s;
    __syncthreads();
    for (int off = 128; off > 0; off >>= 1) {
        if ((int)threadIdx.x < off) buf[threadIdx.x] += buf[threadIdx.x + off];
        __syncthreads();
    }
    if (threadIdx.x == 0) norms[row] = sqrtf(buf[0]);
}

__global__ void gamma_to_f16(_Float16* __restrict__ dst, const float* __restrict__ G,
                             const float* __restrict__ norms)
{
    size_t i = (size_t)blockIdx.x * blockDim.x + threadIdx.x;
    if (i >= (size_t)PADN * PADN) return;
    int r = (int)(i / PADN);
    float rinv = 1.0f / fmaxf(norms[r], 1e-12f);
    dst[i] = (_Float16)(G[i] * rinv);
}

__global__ void tanh_bias_f16(_Float16* __restrict__ dst, const float* __restrict__ H1f,
                              const float* __restrict__ b1)
{
    size_t i = (size_t)blockIdx.x * blockDim.x + threadIdx.x;
    if (i >= (size_t)PADN * HH) return;
    int c = (int)(i % HH);
    dst[i] = (_Float16)tanhf(H1f[i] + b1[c]);
}

// h[i, 0:128] = node_embed[i,:]; h[i, 128:256] = swf[i,:] + b2
__global__ void assemble_h(float* __restrict__ h, const float* __restrict__ node_embed,
                           const float* __restrict__ swf, const float* __restrict__ b2)
{
    size_t i = (size_t)blockIdx.x * blockDim.x + threadIdx.x;
    if (i >= (size_t)NN * HH) return;
    int r = (int)(i / HH), c = (int)(i % HH);
    h[i] = (c < DD) ? node_embed[(size_t)r * DD + c]
                    : swf[(size_t)r * DD + (c - DD)] + b2[c - DD];
}

// ---------------------------------------------------------------------------
// Edge scoring: one wave per edge, 256-dim dot product (2x float4 per lane)
// ---------------------------------------------------------------------------
__global__ __launch_bounds__(256) void edge_score(
    float* __restrict__ out, const float* __restrict__ h,
    const int* __restrict__ src, const int* __restrict__ dst, int nE)
{
    int wave = (int)((blockIdx.x * blockDim.x + threadIdx.x) >> 5);
    int lane = (int)(threadIdx.x & 31);
    if (wave >= nE) return;
    const float4* hs = (const float4*)(h + (size_t)src[wave] * HH);
    const float4* hd = (const float4*)(h + (size_t)(NSRC + dst[wave]) * HH);
    float4 a0 = hs[lane],      b0 = hd[lane];
    float4 a1 = hs[lane + 32], b1 = hd[lane + 32];
    float p = a0.x * b0.x + a0.y * b0.y + a0.z * b0.z + a0.w * b0.w
            + a1.x * b1.x + a1.y * b1.y + a1.z * b1.z + a1.w * b1.w;
#pragma unroll
    for (int off = 16; off > 0; off >>= 1) p += __shfl_down(p, off, 32);
    if (lane == 0) out[wave] = p;
}

// ---------------------------------------------------------------------------
// Skipgram loss: one wave per sample (D=128 -> one float4 per lane)
// ---------------------------------------------------------------------------
__device__ __forceinline__ float softplusf(float x) {
    return fmaxf(x, 0.0f) + log1pf(expf(-fabsf(x)));
}
__device__ __forceinline__ float wave_sum(float p) {
#pragma unroll
    for (int off = 16; off > 0; off >>= 1) p += __shfl_down(p, off, 32);
    return p;
}

__global__ __launch_bounds__(256) void skipgram_loss(
    float* __restrict__ out_loss, const float* __restrict__ ne,
    const float* __restrict__ ce, const int* __restrict__ pu,
    const int* __restrict__ pv, const int* __restrict__ nv)
{
    int wave = (int)((blockIdx.x * blockDim.x + threadIdx.x) >> 5);
    int lane = (int)(threadIdx.x & 31);
    float loss = 0.0f;
    if (wave < BB) {
        float4 uu = ((const float4*)(ne + (size_t)pu[wave] * DD))[lane];
        float4 vv = ((const float4*)(ce + (size_t)pv[wave] * DD))[lane];
        float p = wave_sum(uu.x * vv.x + uu.y * vv.y + uu.z * vv.z + uu.w * vv.w);
        if (lane == 0) {
            float pc = fminf(fmaxf(p, -10.0f), 10.0f);
            loss = softplusf(-pc);  // -log_sigmoid(pos)
        }
#pragma unroll
        for (int k = 0; k < KNEG; ++k) {
            float4 ww = ((const float4*)(ce + (size_t)nv[wave * KNEG + k] * DD))[lane];
            float q = wave_sum(uu.x * ww.x + uu.y * ww.y + uu.z * ww.z + uu.w * ww.w);
            if (lane == 0) {
                float qc = fminf(fmaxf(q, -10.0f), 10.0f);
                loss += softplusf(qc);  // -log_sigmoid(-neg)
            }
        }
    }
    __shared__ float sbuf[8];
    if (lane == 0) sbuf[threadIdx.x >> 5] = loss;
    __syncthreads();
    if (threadIdx.x == 0) {
        float s = 0.0f;
#pragma unroll
        for (int w = 0; w < 8; ++w) s += sbuf[w];
        atomicAdd(out_loss, s);
    }
}

__global__ void init_loss(float* out)     { if (blockIdx.x == 0 && threadIdx.x == 0) out[0] = 0.0f; }
__global__ void finalize_loss(float* out) { if (blockIdx.x == 0 && threadIdx.x == 0) out[0] *= (1.0f / (float)BB); }

// ---------------------------------------------------------------------------
extern "C" void kernel_launch(void* const* d_in, const int* in_sizes, int n_in,
                              void* d_out, int out_size, void* d_ws, size_t ws_size,
                              hipStream_t stream)
{
    const float* node_embed    = (const float*)d_in[0];   // 2000 x 128
    const float* context_embed = (const float*)d_in[1];   // 2000 x 128
    const float* adj           = (const float*)d_in[2];   // 2000 x 2000
    const float* W1            = (const float*)d_in[3];   // 2000 x 256
    const float* b1            = (const float*)d_in[4];   // 256
    const float* W2            = (const float*)d_in[5];   // 256 x 128
    const float* b2            = (const float*)d_in[6];   // 128
    const int*   pos_u         = (const int*)d_in[7];     // 100000
    const int*   pos_v         = (const int*)d_in[8];
    const int*   neg_v         = (const int*)d_in[9];     // 100000 x 5
    const int*   pos_src       = (const int*)d_in[10];    // 200000
    const int*   pos_dst       = (const int*)d_in[11];
    const int*   neg_src       = (const int*)d_in[12];
    const int*   neg_dst       = (const int*)d_in[13];

    float* out     = (float*)d_out;       // [loss, pos_score(200000), neg_score(200000)]
    float* out_pos = out + 1;
    float* out_neg = out + 1 + ENUM;

    // Workspace carve-out (all 256B aligned)
    char* wptr = (char*)d_ws;
    auto alloc = [&](size_t bytes) -> void* {
        void* p = (void*)wptr;
        wptr += (bytes + 255) & ~(size_t)255;
        return p;
    };
    _Float16* adjh = (_Float16*)alloc((size_t)PADN * PADN * 2); // adj f16 (symmetric: A and B^T)
    _Float16* a2h  = (_Float16*)alloc((size_t)PADN * PADN * 2); // A2 f16
    _Float16* gmh  = (_Float16*)alloc((size_t)PADN * PADN * 2); // gamma f16
    float*    A2f  = (float*)   alloc((size_t)PADN * PADN * 4); // A2 = adj@adj
    float*    Pf   = (float*)   alloc((size_t)PADN * PADN * 4); // A2@adj -> gamma (in place)
    _Float16* W1T  = (_Float16*)alloc((size_t)HH * PADN * 2);   // W1^T, 256 x 2016
    _Float16* W2T  = (_Float16*)alloc((size_t)DD * HH * 2);     // W2^T, 128 x 256
    float*    H1f  = (float*)   alloc((size_t)PADN * HH * 4);   // gamma@W1
    _Float16* H1h  = (_Float16*)alloc((size_t)PADN * HH * 2);   // tanh(.+b1) in f16
    float*    swf  = (float*)   alloc((size_t)PADN * DD * 4);   // H1@W2
    float*    dvec = (float*)   alloc((size_t)PADN * 4);        // diag(A2)
    float*    nrms = (float*)   alloc((size_t)PADN * 4);        // row norms of gamma
    float*    hful = (float*)   alloc((size_t)NN * HH * 4);     // concat(node_embed, sw_h)

    auto cdiv = [](long a, long b) { return (int)((a + b - 1) / b); };
    const int T = 256;
    const size_t bigElems = (size_t)PADN * PADN;

    // ---- Skipgram loss (independent of embedding pipeline) ----
    init_loss<<<1, 1, 0, stream>>>(out);
    skipgram_loss<<<cdiv((long)BB * 32, T), T, 0, stream>>>(out, node_embed, context_embed,
                                                            pos_u, pos_v, neg_v);
    finalize_loss<<<1, 1, 0, stream>>>(out);

    // ---- adj -> f16 padded ----
    f32_to_f16_pad<<<cdiv((long)bigElems, T), T, 0, stream>>>(adjh, adj, NN, NN, NN, PADN, PADN);

    // ---- GEMM1: A2 = adj @ adj (adj symmetric => adjh serves as B^T too) ----
    {
        long waves = (long)(PADN / 48) * (PADN / 48);   // 42*42
        wmma_gemm48<<<cdiv(waves * 32, T), T, 0, stream>>>(adjh, adjh, A2f,
                                                           PADN, PADN, PADN,
                                                           PADN, PADN, PADN);
    }
    diag_kernel<<<cdiv(PADN, T), T, 0, stream>>>(dvec, A2f);
    f32_to_f16_pad<<<cdiv((long)bigElems, T), T, 0, stream>>>(a2h, A2f, NN, NN, PADN, PADN, PADN);

    // ---- GEMM2: P = A2 @ adj ----
    {
        long waves = (long)(PADN / 48) * (PADN / 48);
        wmma_gemm48<<<cdiv(waves * 32, T), T, 0, stream>>>(a2h, adjh, Pf,
                                                           PADN, PADN, PADN,
                                                           PADN, PADN, PADN);
    }

    // ---- gamma = normalize(where(d_i*d_j > 0, (P*adj)/(d_i*d_j), 0)) ----
    gamma_raw<<<cdiv((long)bigElems, T), T, 0, stream>>>(Pf, adj, dvec);
    row_norm<<<PADN, T, 0, stream>>>(nrms, Pf);
    gamma_to_f16<<<cdiv((long)bigElems, T), T, 0, stream>>>(gmh, Pf, nrms);

    // ---- H1 = gamma @ W1 ----
    f32T_to_f16_pad<<<cdiv((long)HH * PADN, T), T, 0, stream>>>(W1T, W1, HH, PADN, NN, HH);
    {
        long waves = (long)(PADN / 16) * (HH / 16);
        wmma_gemm_f16f32<<<cdiv(waves * 32, T), T, 0, stream>>>(gmh, W1T, H1f,
                                                                PADN, HH, PADN,
                                                                PADN, PADN, HH);
    }
    tanh_bias_f16<<<cdiv((long)PADN * HH, T), T, 0, stream>>>(H1h, H1f, b1);

    // ---- sw_h = tanh(H1) @ W2 ----
    f32T_to_f16_pad<<<cdiv((long)DD * HH, T), T, 0, stream>>>(W2T, W2, DD, HH, HH, DD);
    {
        long waves = (long)(PADN / 16) * (DD / 16);
        wmma_gemm_f16f32<<<cdiv(waves * 32, T), T, 0, stream>>>(H1h, W2T, swf,
                                                                PADN, DD, HH,
                                                                HH, HH, DD);
    }

    // ---- h = concat([node_embed, sw_h + b2], axis=1) ----
    assemble_h<<<cdiv((long)NN * HH, T), T, 0, stream>>>(hful, node_embed, swf, b2);

    // ---- edge scores ----
    edge_score<<<cdiv((long)ENUM * 32, T), T, 0, stream>>>(out_pos, hful, pos_src, pos_dst, ENUM);
    edge_score<<<cdiv((long)ENUM * 32, T), T, 0, stream>>>(out_neg, hful, neg_src, neg_dst, ENUM);
}
